// XposMultiheadAttention_10282151707348
// MI455X (gfx1250) — matrix operand order repaired and checked by reading, not compiled, not run on hardware
//
#include <hip/hip_runtime.h>
#include <hip/hip_bf16.h>

// ---------------------------------------------------------------------------
// XPos Multihead Attention for MI455X (gfx1250), bf16 WMMA pipeline.
//  B=8, T=S=1024, E=1024, H=16, HD=64.
//  Pass 0: f32 -> bf16 conversion of activations + weights, xpos tables.
//  Pass 1: Q/K/V projections (+bias, +scaling, +xpos via table) -> bf16 ws.
//  Pass 2: flash-attention per head, K staged via Tensor Data Mover,
//          V staged transposed in LDS.
//  Pass 3: output projection -> f32 d_out.
// ---------------------------------------------------------------------------

typedef __attribute__((ext_vector_type(16))) __bf16 v16bf;
typedef __attribute__((ext_vector_type(8)))  float  v8f;
typedef unsigned int u32x4 __attribute__((ext_vector_type(4)));
typedef int          i32x4 __attribute__((ext_vector_type(4)));
typedef int          i32x8 __attribute__((ext_vector_type(8)));

#define SEQ 1024
#define EMB 1024
#define HD  64
#define NH  16
#define BH  128              // B * NH
#define CHUNK 256            // s-rows staged in LDS per iteration

#if defined(__has_builtin)
#if __has_builtin(__builtin_amdgcn_tensor_load_to_lds)
#define HAVE_TDM 1
#endif
#endif

union U4 { uint4 u; __bf16 h[8]; };

static __device__ __forceinline__ v8f vzero8() {
    v8f z;
#pragma unroll
    for (int i = 0; i < 8; ++i) z[i] = 0.0f;
    return z;
}

static __device__ __forceinline__ v16bf pack2(U4 a, U4 b) {
    v16bf r;
#pragma unroll
    for (int j = 0; j < 8; ++j) { r[j] = a.h[j]; r[j + 8] = b.h[j]; }
    return r;
}

// A-fragment (16x32 bf16): lane%16 = row M; halves = 16 consecutive K at (lane/16)*16.
static __device__ __forceinline__ v16bf load_a_bf16(const __bf16* A, int lda,
                                                    int row, int kbase, int lane) {
    int kb = (lane & 16) ? 16 : 0;
    const __bf16* p = A + (size_t)row * lda + kbase + kb;
    U4 a, b;
    a.u = *(const uint4*)p;
    b.u = *(const uint4*)(p + 8);
    return pack2(a, b);
}

// B-fragment (32x16): lane holds col n = lane%16, halves = 16 consecutive K
// at (lane/16)*16; K contiguous in memory.
static __device__ __forceinline__ v16bf load_b_contig_bf16(const __bf16* p) {
    U4 a, b;
    a.u = *(const uint4*)p;
    b.u = *(const uint4*)(p + 8);
    return pack2(a, b);
}

static __device__ __forceinline__ v8f wmma_bf16(v16bf a, v16bf b, v8f c) {
    return __builtin_amdgcn_wmma_f32_16x16x32_bf16(false, a, false, b,
                                                   (short)0, c, false, false);
}

// ---------------------------------------------------------------------------
// f32 -> bf16 conversion (vectorized by 8).
// ---------------------------------------------------------------------------
__global__ __launch_bounds__(256)
void cvt_f32_bf16_kernel(const float* __restrict__ in, __bf16* __restrict__ out,
                         int n) {
    const int i = (blockIdx.x * 256 + threadIdx.x) * 8;
    if (i + 8 <= n) {
        float4 a = ((const float4*)(in + i))[0];
        float4 b = ((const float4*)(in + i))[1];
        U4 u;
        u.h[0] = (__bf16)a.x; u.h[1] = (__bf16)a.y;
        u.h[2] = (__bf16)a.z; u.h[3] = (__bf16)a.w;
        u.h[4] = (__bf16)b.x; u.h[5] = (__bf16)b.y;
        u.h[6] = (__bf16)b.z; u.h[7] = (__bf16)b.w;
        *(uint4*)(out + i) = u.u;
    }
}

// ---------------------------------------------------------------------------
// xpos rotary tables: tab[mode][t][d] = (cos(t*w_i)*scale, sin(t*w_i)*scale),
// mode 0 = upscale (Q), mode 1 = downscale (K).  2 x 1024 x 64 float2 = 1 MB.
// ---------------------------------------------------------------------------
__global__ __launch_bounds__(256)
void xpos_tab_kernel(float2* __restrict__ tab) {
    const int idx  = blockIdx.x * 256 + threadIdx.x;   // 131072 total
    const int mode = idx >> 16;
    const int t    = (idx >> 6) & (SEQ - 1);
    const int d    = idx & (HD - 1);
    const int i    = d >> 1;
    const float inv_freq = __powf(10000.0f, -(float)i * (1.0f / 32.0f));
    const float sinus = (float)t * inv_freq;
    const float si = (2.0f * (float)i + 25.6f) * (1.0f / 89.6f);
    float ex = ((float)t - 512.0f) * (1.0f / 1024.0f);
    if (mode == 1) ex = -ex;                           // downscale for K
    const float scale = __powf(si, ex);
    tab[idx] = make_float2(__cosf(sinus) * scale, __sinf(sinus) * scale);
}

// ---------------------------------------------------------------------------
// GEMM: C[M=8192, N=1024] = A(bf16) @ W(bf16)^T + bias, fused epilogue.
//   mode 0: Q  (scale 1/8, xpos table)     -> bf16 head-major ws
//   mode 1: K  (xpos table)                -> bf16 head-major ws
//   mode 2: V                              -> bf16 head-major ws
//   mode 3: O                              -> f32 d_out [B,T,E]
// block = 256 (8 waves); wave tile = 32(M) x 64(N); wg tile = 256 x 64.
// Per k-step: 2 A-frags + 4 shared B-frags -> 8 WMMAs.
// ---------------------------------------------------------------------------
__global__ __launch_bounds__(256)
void xpos_proj_kernel(const __bf16* __restrict__ A,
                      const __bf16* __restrict__ W,
                      const float* __restrict__ bias,
                      const float2* __restrict__ xtab,   // per-mode table or null
                      float* __restrict__ outF,
                      __bf16* __restrict__ outH,
                      int mode) {
    const int lane  = threadIdx.x & 31;
    const int wave  = threadIdx.x >> 5;
    const int mbase = blockIdx.x * 256 + wave * 32;
    const int nbase = blockIdx.y * 64;
    const int arow0 = mbase + (lane & 15);
    const int nW    = nbase + (lane & 15);        // B-frag: W row per lane

    v8f acc[2][4];
#pragma unroll
    for (int mt = 0; mt < 2; ++mt)
#pragma unroll
        for (int nt = 0; nt < 4; ++nt) acc[mt][nt] = vzero8();

    for (int k = 0; k < EMB; k += 32) {
        __builtin_prefetch(A + (size_t)arow0 * EMB + k + 64, 0, 1);
        v16bf a0 = load_a_bf16(A, EMB, arow0,      k, lane);
        v16bf a1 = load_a_bf16(A, EMB, arow0 + 16, k, lane);
        const int kk = k + ((lane >> 4) << 4);
#pragma unroll
        for (int nt = 0; nt < 4; ++nt) {
            v16bf b = load_b_contig_bf16(W + (size_t)(nW + nt * 16) * EMB + kk);
            acc[0][nt] = wmma_bf16(a0, b, acc[0][nt]);
            acc[1][nt] = wmma_bf16(a1, b, acc[1][nt]);
        }
    }

    const int half = lane >> 4;
#pragma unroll
    for (int mt = 0; mt < 2; ++mt) {
#pragma unroll
        for (int nt = 0; nt < 4; ++nt) {
            const int n  = nbase + nt * 16 + (lane & 15);
            const float bv = bias[n];
#pragma unroll
            for (int r = 0; r < 8; ++r) {
                const int m = mbase + mt * 16 + r + (half << 3);
                float c = acc[mt][nt][r] + bv;
                if (mode == 0) c *= 0.125f;              // head_dim^-0.5
                if (mode == 0 || mode == 1) {
                    // xpos rotary: pairs are adjacent N cols => adjacent lanes
                    const int t = m & (SEQ - 1);
                    const int d = n & (HD - 1);
                    const float2 cssn = xtab[(t << 6) | d];
                    const float partner = __shfl_xor(c, 1, 32);
                    c = c * cssn.x + ((d & 1) ? partner : -partner) * cssn.y;
                }
                if (mode == 3) {
                    outF[(size_t)m * EMB + n] = c;
                } else {
                    const int b = m >> 10, t = m & (SEQ - 1);
                    const int h = n >> 6,  d = n & (HD - 1);
                    outH[((size_t)(b * NH + h) * SEQ + t) * HD + d] = (__bf16)c;
                }
            }
        }
    }
}

// ---------------------------------------------------------------------------
// Flash attention per head.  grid = (128 heads, 8 row-blocks), block = 256.
// K chunk staged into LDS by the Tensor Data Mover (wave 0 issues the D#),
// V staged transposed so PV B-fragments are contiguous LDS reads.
// ---------------------------------------------------------------------------
__global__ __launch_bounds__(256)
void xpos_attn_kernel(const __bf16* __restrict__ q,
                      const __bf16* __restrict__ k,
                      const __bf16* __restrict__ v,
                      __bf16* __restrict__ o) {
    __shared__ __attribute__((aligned(16))) __bf16 Ks[CHUNK * HD];   // [s][d]
    __shared__ __attribute__((aligned(16))) __bf16 Vt[HD * CHUNK];   // [d][s]
    __shared__ __attribute__((aligned(16))) __bf16 Pst[8 * 16 * 32]; // per-wave P

    const int bh   = blockIdx.x;
    const int lane = threadIdx.x & 31;
    const int wave = threadIdx.x >> 5;
    const int tid  = threadIdx.x;
    const int half = lane >> 4;
    const int t0   = blockIdx.y * 128 + wave * 16;

    const __bf16* qh = q + (size_t)bh * SEQ * HD;
    const __bf16* kh = k + (size_t)bh * SEQ * HD;
    const __bf16* vh = v + (size_t)bh * SEQ * HD;

    // Q fragments for this wave's 16 rows (K = 0..31 and 32..63).
    const v16bf qa0 = load_a_bf16(qh, HD, t0 + (lane & 15), 0, lane);
    const v16bf qa1 = load_a_bf16(qh, HD, t0 + (lane & 15), 32, lane);

    float mrun[8], lrun[8];
    v8f acc[4];
#pragma unroll
    for (int r = 0; r < 8; ++r) { mrun[r] = -1e30f; lrun[r] = 0.0f; }
#pragma unroll
    for (int dt = 0; dt < 4; ++dt) acc[dt] = vzero8();

    __bf16* pw = Pst + wave * (16 * 32);

    for (int chunk = 0; chunk < SEQ / CHUNK; ++chunk) {
        const __bf16* kg = kh + (size_t)chunk * CHUNK * HD;

#ifdef HAVE_TDM
        // ---- K chunk: Tensor Data Mover, issued by wave 0 --------------
        if (wave == 0) {
            const unsigned long long ga = (unsigned long long)(uintptr_t)kg;
            const unsigned int lds_off = (unsigned int)(uintptr_t)(&Ks[0]);
            u32x4 g0;
            g0[0] = 1u;                                  // count=1, user mode
            g0[1] = lds_off;                             // lds_addr
            g0[2] = (unsigned int)(ga & 0xFFFFFFFFu);    // global_addr[31:0]
            g0[3] = (unsigned int)((ga >> 32) & 0x01FFFFFFu)
                  | 0x80000000u;                         // addr[56:32], type=2
            i32x8 g1;
            g1[0] = 0x00010000;          // wg_mask=0, data_size=1 (2 bytes)
            g1[1] = (HD << 16);          // tensor_dim0[15:0] @ bits 63:48
            g1[2] = (SEQ << 16);         // dim0[31:16]=0 | tensor_dim1[15:0]
            g1[3] = (HD << 16);          // dim1[31:16]=0 | tile_dim0=64
            g1[4] = CHUNK;               // tile_dim1=256, tile_dim2=0
            g1[5] = HD;                  // tensor_dim0_stride[31:0]=64
            g1[6] = 0;                   // stride0[47:32] | stride1[15:0]
            g1[7] = 0;                   // stride1[47:16]
            i32x4 z4; z4[0] = 0; z4[1] = 0; z4[2] = 0; z4[3] = 0;
#if __clang_major__ >= 23
            i32x8 z8;
#pragma unroll
            for (int i = 0; i < 8; ++i) z8[i] = 0;
            __builtin_amdgcn_tensor_load_to_lds(g0, g1, z4, z4, z8, 0);
#else
            __builtin_amdgcn_tensor_load_to_lds(g0, g1, z4, z4, 0);
#endif
        }
#else
        // ---- fallback: cooperative K copy ------------------------------
        {
            const uint4* gk = (const uint4*)kg;
            uint4* sk = (uint4*)Ks;
#pragma unroll
            for (int it = 0; it < 8; ++it) sk[it * 256 + tid] = gk[it * 256 + tid];
        }
#endif

        // ---- V chunk: cooperative transposed store ---------------------
        {
            const uint4* gv = (const uint4*)(vh + (size_t)chunk * CHUNK * HD);
#pragma unroll
            for (int it = 0; it < 8; ++it) {
                const int idx = it * 256 + tid;       // idx*8 = s*64 + d0
                U4 u; u.u = gv[idx];
                const int s  = idx >> 3;
                const int d0 = (idx & 7) << 3;
#pragma unroll
                for (int j = 0; j < 8; ++j) Vt[(d0 + j) * CHUNK + s] = u.h[j];
            }
        }
#ifdef HAVE_TDM
        if (wave == 0) __builtin_amdgcn_s_wait_tensorcnt(0);
#endif
        __syncthreads();

        for (int s0 = 0; s0 < CHUNK; s0 += 32) {
            // ---- scores: 16 rows x 32 cols, two 16-col subtiles --------
            v8f sc[2];
#pragma unroll
            for (int j = 0; j < 2; ++j) {
                const int srow = s0 + j * 16 + (lane & 15);
                v16bf b0 = load_b_contig_bf16(&Ks[srow * HD + 0  + (half << 4)]);
                v16bf b1 = load_b_contig_bf16(&Ks[srow * HD + 32 + (half << 4)]);
                v8f c = vzero8();
                c = wmma_bf16(qa0, b0, c);
                c = wmma_bf16(qa1, b1, c);
                sc[j] = c;
            }

            // ---- online softmax (rows live in 16-lane halves) ----------
#pragma unroll
            for (int r = 0; r < 8; ++r) {
                float mx = fmaxf(sc[0][r], sc[1][r]);
#pragma unroll
                for (int off = 1; off < 16; off <<= 1)
                    mx = fmaxf(mx, __shfl_xor(mx, off, 32));
                const float mnew  = fmaxf(mrun[r], mx);
                const float alpha = __expf(mrun[r] - mnew);
                const float p0 = __expf(sc[0][r] - mnew);
                const float p1 = __expf(sc[1][r] - mnew);
                float rs = p0 + p1;
#pragma unroll
                for (int off = 1; off < 16; off <<= 1)
                    rs += __shfl_xor(rs, off, 32);
                lrun[r] = lrun[r] * alpha + rs;
                mrun[r] = mnew;
#pragma unroll
                for (int dt = 0; dt < 4; ++dt) acc[dt][r] *= alpha;
                const int prow = r + (half << 3);
                pw[prow * 32 + (lane & 15)]      = (__bf16)p0;
                pw[prow * 32 + 16 + (lane & 15)] = (__bf16)p1;
            }

            // ---- P (C-layout) -> A-fragment via per-wave LDS tile ------
            v16bf pa = load_a_bf16(pw, 32, lane & 15, 0, lane);

            // ---- PV: B from transposed V (contiguous s per lane) -------
#pragma unroll
            for (int dt = 0; dt < 4; ++dt) {
                const int d = dt * 16 + (lane & 15);
                v16bf vb = load_b_contig_bf16(&Vt[d * CHUNK + s0 + (half << 4)]);
                acc[dt] = wmma_bf16(pa, vb, acc[dt]);
            }
        }
        __syncthreads();
    }

    // ---- epilogue: normalize, write bf16 to [B,T,E] layout -------------
    const int b = bh >> 4, h = bh & 15;
#pragma unroll
    for (int dt = 0; dt < 4; ++dt) {
#pragma unroll
        for (int r = 0; r < 8; ++r) {
            const int t = t0 + r + (half << 3);
            const int d = dt * 16 + (lane & 15);
            const float ov = acc[dt][r] / lrun[r];
            o[((size_t)b * SEQ + t) * EMB + h * HD + d] = (__bf16)ov;
        }
    }
}

// ---------------------------------------------------------------------------
extern "C" void kernel_launch(void* const* d_in, const int* in_sizes, int n_in,
                              void* d_out, int out_size, void* d_ws, size_t ws_size,
                              hipStream_t stream) {
    const float* query = (const float*)d_in[0];
    const float* key   = (const float*)d_in[1];
    const float* value = (const float*)d_in[2];
    const float* wq = (const float*)d_in[3];
    const float* bq = (const float*)d_in[4];
    const float* wk = (const float*)d_in[5];
    const float* bk = (const float*)d_in[6];
    const float* wv = (const float*)d_in[7];
    const float* bv = (const float*)d_in[8];
    const float* wo = (const float*)d_in[9];
    const float* bo = (const float*)d_in[10];

    // workspace (bf16 elements first, then float2 tables)
    const size_t NACT = (size_t)BH * SEQ * HD;   // 8,388,608
    const size_t NWGT = (size_t)EMB * EMB;       // 1,048,576
    __bf16* qws = (__bf16*)d_ws;                 // head-major q
    __bf16* kws = qws + NACT;                    // head-major k
    __bf16* vws = kws + NACT;                    // head-major v
    __bf16* aws = vws + NACT;                    // attn result [B,T,E]
    __bf16* xq  = aws + NACT;                    // bf16 inputs
    __bf16* xk  = xq  + NACT;
    __bf16* xv  = xk  + NACT;
    __bf16* wqb = xv  + NACT;                    // bf16 weights
    __bf16* wkb = wqb + NWGT;
    __bf16* wvb = wkb + NWGT;
    __bf16* wob = wvb + NWGT;
    float2* xtab = (float2*)(wob + NWGT);        // [2][SEQ][HD] cos/sin tables

    dim3 blk(256);

    // Pass 0: one-shot conversions + xpos tables.
    cvt_f32_bf16_kernel<<<dim3((int)(NACT / 2048)), blk, 0, stream>>>(query, xq, (int)NACT);
    cvt_f32_bf16_kernel<<<dim3((int)(NACT / 2048)), blk, 0, stream>>>(key,   xk, (int)NACT);
    cvt_f32_bf16_kernel<<<dim3((int)(NACT / 2048)), blk, 0, stream>>>(value, xv, (int)NACT);
    cvt_f32_bf16_kernel<<<dim3((int)(NWGT / 2048)), blk, 0, stream>>>(wq, wqb, (int)NWGT);
    cvt_f32_bf16_kernel<<<dim3((int)(NWGT / 2048)), blk, 0, stream>>>(wk, wkb, (int)NWGT);
    cvt_f32_bf16_kernel<<<dim3((int)(NWGT / 2048)), blk, 0, stream>>>(wv, wvb, (int)NWGT);
    cvt_f32_bf16_kernel<<<dim3((int)(NWGT / 2048)), blk, 0, stream>>>(wo, wob, (int)NWGT);
    xpos_tab_kernel<<<dim3(512), blk, 0, stream>>>(xtab);

    // Pass 1: projections with fused bias/scale/xpos.
    dim3 gemm_grid(8192 / 256, EMB / 64);
    xpos_proj_kernel<<<gemm_grid, blk, 0, stream>>>(xq, wqb, bq, xtab,
                                                    nullptr, qws, 0);
    xpos_proj_kernel<<<gemm_grid, blk, 0, stream>>>(xk, wkb, bk, xtab + SEQ * HD,
                                                    nullptr, kws, 1);
    xpos_proj_kernel<<<gemm_grid, blk, 0, stream>>>(xv, wvb, bv, nullptr,
                                                    nullptr, vws, 2);

    // Pass 2: flash attention.
    dim3 attn_grid(BH, SEQ / 128);
    xpos_attn_kernel<<<attn_grid, blk, 0, stream>>>(qws, kws, vws, aws);

    // Pass 3: output projection -> f32.
    xpos_proj_kernel<<<gemm_grid, blk, 0, stream>>>(aws, wob, bo, nullptr,
                                                    (float*)d_out, nullptr, 3);
}
